// AgnosticChargeBiasedLinearPotentialEmbedding_86835648790769
// MI455X (gfx1250) — matrix operand order repaired and checked by reading, not compiled, not run on hardware
//
#include <hip/hip_runtime.h>

// CDNA5 / gfx1250, wave32. bf16x3-split WMMA GEMM for the e3nn irrep linear.
typedef __attribute__((ext_vector_type(16))) __bf16 v16bf;
typedef __attribute__((ext_vector_type(8)))  float  v8f;

union AFrag { v16bf v; unsigned u[8]; };
union BFrag { v16bf v; uint4 q[2]; };

static __device__ __forceinline__ unsigned short bf16_rne(float f) {
  unsigned u = __float_as_uint(f);
  u += 0x7FFFu + ((u >> 16) & 1u);          // round-to-nearest-even to bf16
  return (unsigned short)(u >> 16);
}
static __device__ __forceinline__ float bf16_hi_to_f32(unsigned short h) {
  return __uint_as_float(((unsigned)h) << 16);
}

// Pack the high 16 bits of two f32 bit patterns into one dword:
// result = {u1[31:16], u0[31:16]}  (= two truncated bf16, low component first).
static __device__ __forceinline__ unsigned pack_hi16(unsigned u0, unsigned u1) {
#if __has_builtin(__builtin_amdgcn_perm)
  return __builtin_amdgcn_perm(u1, u0, 0x07060302u);  // 1x v_perm_b32
#else
  return (u0 >> 16) | (u1 & 0xFFFF0000u);
#endif
}

template <int L>
__global__ __launch_bounds__(512)
void acblpe_kernel(const float* __restrict__ pot,    // [n,9]
                   const float* __restrict__ node,   // [n,1152]
                   const float* __restrict__ chg,    // [n,1]
                   const float* __restrict__ Wp,     // [128]  (Wp_L)
                   const float* __restrict__ Wn,     // [128,128] (Wn_L)
                   const float* __restrict__ Wc0,    // [128]
                   float* __restrict__ out,          // [n,1152]
                   int n)
{
  constexpr int w      = 2 * L + 1;                 // irrep dim, compile-time
  constexpr int segOff = (L == 0) ? 0 : (L == 1 ? 128 : 512);
  constexpr int potOff = L * L;

  // 64 KB LDS: W_L as bf16 hi (32KB) | bf16 lo (32KB), pre-swizzled into
  // WMMA B-operand fragment order (lane: N=lane&15, K=kbase+(lane>>4)*8+t+(t>=8?8:0)).
  // Weights use a full RNE split (done once per block): |B_lo| <= 2^-9 |B|.
  __shared__ unsigned short ldsW[2 * 32 * 512];

  const int tid = threadIdx.x;
  for (int e = tid; e < 16384; e += 512) {
    int i = e >> 7, o = e & 127;                    // W[i(=k), o]
    float wv = Wn[e];
    int kk = i >> 5, kp = i & 31;
    int ln = (((kp >> 3) & 1) << 4) | (o & 15);
    int idx = ((o >> 4) * 4 + kk) * 512 + ln * 16 + ((kp & 7) | ((kp >> 4) << 3));
    unsigned short hb = bf16_rne(wv);
    ldsW[idx]         = hb;
    ldsW[16384 + idx] = bf16_rne(wv - bf16_hi_to_f32(hb));
  }
  __syncthreads();

  // ---- wave -> 16-row stripe of this segment's GEMM ----
  const int segRows    = n * w;
  const int segStripes = (segRows + 15) >> 4;
  const int stripe     = (int)blockIdx.x * 16 + (tid >> 5);
  if (stripe >= segStripes) return;                 // wave-uniform: EXEC stays full

  const int row0 = stripe << 4;
  const int lane = tid & 31;
  const int h    = lane >> 4;
  const int lm   = lane & 15;

  // A 16x32 bf16 layout: lane holds M = lane&15; comps 0..7 -> K = h*8+j,
  // comps 8..15 -> K = 16+h*8+j (two bf16 per VGPR dword).
  int rA = row0 + lm;
  if (rA >= segRows) rA = segRows - 1;              // clamp (keeps EXEC full)
  const int nA = rA / w, mA = rA - nA * w;          // w constexpr -> magic mul
  const unsigned aoff = (unsigned)nA * 1152u + (unsigned)segOff + (unsigned)mA;

  v8f zero = {0.f, 0.f, 0.f, 0.f, 0.f, 0.f, 0.f, 0.f};
  v8f acc[8];
  #pragma unroll
  for (int c = 0; c < 8; ++c) acc[c] = zero;

  #pragma unroll
  for (int kk = 0; kk < 4; ++kk) {
    // A fragment via truncation split: hi = top16(x) (exact residual),
    // lo = trunc16(x - hi). ~1.5 VALU/element; overall error ~2^-16.
    AFrag ahi, alo;
    const unsigned ab = aoff + (unsigned)((kk * 32 + h * 8) * w);
    #pragma unroll
    for (int d = 0; d < 4; ++d) {
      float x0 = node[ab + (2 * d) * w];
      float x1 = node[ab + (2 * d + 1) * w];
      float y0 = node[ab + (16 + 2 * d) * w];
      float y1 = node[ab + (17 + 2 * d) * w];
      unsigned u0 = __float_as_uint(x0), u1 = __float_as_uint(x1);
      unsigned p0 = __float_as_uint(y0), p1 = __float_as_uint(y1);
      ahi.u[d]     = pack_hi16(u0, u1);
      ahi.u[4 + d] = pack_hi16(p0, p1);
      float l0 = x0 - __uint_as_float(u0 & 0xFFFF0000u);   // exact
      float l1 = x1 - __uint_as_float(u1 & 0xFFFF0000u);   // exact
      float l2 = y0 - __uint_as_float(p0 & 0xFFFF0000u);   // exact
      float l3 = y1 - __uint_as_float(p1 & 0xFFFF0000u);   // exact
      alo.u[d]     = pack_hi16(__float_as_uint(l0), __float_as_uint(l1));
      alo.u[4 + d] = pack_hi16(__float_as_uint(l2), __float_as_uint(l3));
    }
    #pragma unroll
    for (int ct = 0; ct < 8; ++ct) {
      const unsigned short* bp = &ldsW[(ct * 4 + kk) * 512 + lane * 16];
      BFrag bhi, blo;
      bhi.q[0] = *(const uint4*)(bp);
      bhi.q[1] = *(const uint4*)(bp + 8);
      blo.q[0] = *(const uint4*)(bp + 16384);
      blo.q[1] = *(const uint4*)(bp + 16384 + 8);
      // bf16x3: A_hi*B_hi + A_lo*B_hi + A_hi*B_lo (~fp32-grade, f32 accum)
      acc[ct] = __builtin_amdgcn_wmma_f32_16x16x32_bf16(false, ahi.v, false, bhi.v,
                                                        (short)0, acc[ct], false, false);
      acc[ct] = __builtin_amdgcn_wmma_f32_16x16x32_bf16(false, alo.v, false, bhi.v,
                                                        (short)0, acc[ct], false, false);
      acc[ct] = __builtin_amdgcn_wmma_f32_16x16x32_bf16(false, ahi.v, false, blo.v,
                                                        (short)0, acc[ct], false, false);
    }
  }

  // ---- fused epilogue: /sqrt(128) + rank-1 pot/charge embeddings ----
  // C/D layout: lane, VGPR v -> M = v + 8*(lane>>4), N = lane&15.
  const float scale = 0.08838834764831845f;         // 1/sqrt(128)
  float wpv[8], wcv[8];
  #pragma unroll
  for (int ct = 0; ct < 8; ++ct) {
    wpv[ct] = Wp[ct * 16 + lm];
    if constexpr (L == 0) wcv[ct] = Wc0[ct * 16 + lm]; else wcv[ct] = 0.f;
  }
  const unsigned lmw = (unsigned)(lm * w);
  int r  = row0 + h * 8;
  int nn = r / w, mm = r - nn * w;
  #pragma unroll
  for (int v = 0; v < 8; ++v) {
    if (r < segRows) {
      const float pv = pot[(unsigned)nn * 9u + potOff + (unsigned)mm];
      const unsigned ob = (unsigned)nn * 1152u + segOff + (unsigned)mm + lmw;
      if constexpr (L == 0) {
        const float cg = chg[nn];
        #pragma unroll
        for (int ct = 0; ct < 8; ++ct)
          out[ob + ct * 16 * w] =
              fmaf(acc[ct][v], scale, fmaf(wpv[ct], pv, wcv[ct] * cg));
      } else {
        #pragma unroll
        for (int ct = 0; ct < 8; ++ct)
          out[ob + ct * 16 * w] = fmaf(acc[ct][v], scale, wpv[ct] * pv);
      }
    }
    ++r;
    if (++mm == w) { mm = 0; ++nn; }
  }
}

extern "C" void kernel_launch(void* const* d_in, const int* in_sizes, int n_in,
                              void* d_out, int out_size, void* d_ws, size_t ws_size,
                              hipStream_t stream) {
  const float* pot  = (const float*)d_in[0];
  const float* node = (const float*)d_in[1];
  // d_in[2] = node_attrs: explicitly unused by the reference.
  const float* chg  = (const float*)d_in[3];
  const float* Wp0  = (const float*)d_in[4];
  const float* Wp1  = (const float*)d_in[5];
  const float* Wp2  = (const float*)d_in[6];
  const float* Wn0  = (const float*)d_in[7];
  const float* Wn1  = (const float*)d_in[8];
  const float* Wn2  = (const float*)d_in[9];
  const float* Wc0  = (const float*)d_in[10];
  float* out = (float*)d_out;

  const int n  = in_sizes[1] / 1152;
  const int s0 = (n * 1 + 15) / 16;                 // 16-row stripes per segment
  const int s1 = (n * 3 + 15) / 16;
  const int s2 = (n * 5 + 15) / 16;

  acblpe_kernel<0><<<dim3((s0 + 15) / 16), dim3(512), 0, stream>>>(
      pot, node, chg, Wp0, Wn0, Wc0, out, n);
  acblpe_kernel<1><<<dim3((s1 + 15) / 16), dim3(512), 0, stream>>>(
      pot, node, chg, Wp1, Wn1, Wc0, out, n);
  acblpe_kernel<2><<<dim3((s2 + 15) / 16), dim3(512), 0, stream>>>(
      pot, node, chg, Wp2, Wn2, Wc0, out, n);
}
